// DiversityLoss_51969104281992
// MI455X (gfx1250) — matrix operand order repaired and checked by reading, not compiled, not run on hardware
//
#include <hip/hip_runtime.h>
#include <hip/hip_bf16.h>
#include <math.h>

// MI455X / gfx1250, wave32. Fused diversity-loss via trace identity:
//   Sum_ij D_img*D_noise = 2B*Sst + 2*Ss*St - 4*Ssrn - 4*Svw + 4*||X^T N||_F^2
// Dominant work: one HBM pass over X (201 MB ~ 8.6us @ 23.3 TB/s) feeding a
// 12288x256x4096 GEMM done with v_wmma_f32_16x16x32_bf16 (hi/lo split -> ~fp32).

typedef __attribute__((ext_vector_type(16))) __bf16 v16bf;
typedef __attribute__((ext_vector_type(8)))  __bf16 v8bf;
typedef __attribute__((ext_vector_type(8)))  float  v8f;
typedef __attribute__((ext_vector_type(4)))  float  v4f;

#define NB 4096      // batch
#define D1 12288     // image feature dim (3*64*64)
#define D2 256       // noise feature dim
#define KT 64        // k-rows of M per workgroup
#define LDSROW 40    // padded bf16 stride of LDS X^T rows (80B, 16B aligned)

// ---- noises: per-row squared norms t[i] ----
__global__ __launch_bounds__(256) void k_noise_rowstats(const float* __restrict__ N,
                                                        float* __restrict__ t_arr) {
  int i = blockIdx.x * 256 + threadIdx.x;
  const float* r = N + (size_t)i * D2;
  float acc = 0.f;
  #pragma unroll 4
  for (int l = 0; l < D2; l += 4) {
    v4f x = *(const v4f*)(r + l);
    acc += x.x * x.x + x.y * x.y + x.z * x.z + x.w * x.w;
  }
  t_arr[i] = acc;
}

// ---- noises: transpose + bf16 hi/lo split: Nt[l][i] ----
__global__ __launch_bounds__(256) void k_noise_split(const float* __restrict__ N,
                                                     __bf16* __restrict__ Nth,
                                                     __bf16* __restrict__ Ntl) {
  unsigned g = blockIdx.x * 256u + threadIdx.x;
  unsigned i = g & (NB - 1);
  unsigned l = g >> 12;           // NB == 2^12
  float x = N[(size_t)i * D2 + l];
  __bf16 h  = (__bf16)x;
  __bf16 lo = (__bf16)(x - (float)h);
  Nth[(size_t)l * NB + i] = h;
  Ntl[(size_t)l * NB + i] = lo;
}

// ---- noises: column sums u[l] = sum_i N[i,l] ----
__global__ __launch_bounds__(256) void k_noise_colsum(const float* __restrict__ N,
                                                      float* __restrict__ u) {
  int l = threadIdx.x;
  int i0 = blockIdx.x * 256;
  float acc = 0.f;
  for (int i = 0; i < 256; ++i) acc += N[(size_t)(i0 + i) * D2 + l];
  atomicAdd(&u[l], acc);
}

// ---- noises: rn[i] = n_i . u ----
__global__ __launch_bounds__(256) void k_noise_rn(const float* __restrict__ N,
                                                  const float* __restrict__ u,
                                                  float* __restrict__ rn) {
  __shared__ float us[D2];
  us[threadIdx.x] = u[threadIdx.x];
  __syncthreads();
  int i = blockIdx.x * 256 + threadIdx.x;
  const float* r = N + (size_t)i * D2;
  float acc = 0.f;
  #pragma unroll 4
  for (int l = 0; l < D2; ++l) acc += r[l] * us[l];
  rn[i] = acc;
}

// ---- main: one pass over X. Computes F = ||X^T N||_F^2 (WMMA, hi/lo split),
//      s[i] += sum_k x^2, and Svw += sum_k v[k]*w[k] for this k-slab. ----
__global__ __launch_bounds__(256) void k_main(const float* __restrict__ X,
                                              const __bf16* __restrict__ Nth,
                                              const __bf16* __restrict__ Ntl,
                                              const float* __restrict__ t_arr,
                                              float* __restrict__ s_arr,
                                              float* __restrict__ scal) {
  __shared__ __bf16 Xh[KT * LDSROW];   // X^T tile, hi part: [k_local][i_local]
  __shared__ __bf16 Xl[KT * LDSROW];   // lo part
  __shared__ float v_l[KT];
  __shared__ float w_l[KT];

  const int tid  = threadIdx.x;
  const int lane = tid & 31;
  const int wave = tid >> 5;           // 8 waves
  const int k0   = blockIdx.x * KT;    // this wg's k-slab of M

  if (tid < KT) { v_l[tid] = 0.f; w_l[tid] = 0.f; }

  // wave -> 8 output tiles: k-tile kt (16 rows), l-tiles lgrp*8 + 0..7
  const int kt   = wave & 3;
  const int lgrp = wave >> 2;
  const int half = lane >> 4;          // ISA 16-bit A layout lane halves
  const int m    = lane & 15;
  const int arow = kt * 16 + m;

  // staging assignment: 8 threads per i-row, 8 consecutive k each
  const int i_loc = tid >> 3;          // 0..31
  const int j8    = tid & 7;
  const int kl    = j8 * 8;

  v8f acc[8];
  #pragma unroll
  for (int j = 0; j < 8; ++j) acc[j] = {};

  float vacc[8], wacc[8];
  #pragma unroll
  for (int q = 0; q < 8; ++q) { vacc[q] = 0.f; wacc[q] = 0.f; }

  for (int i0 = 0; i0 < NB; i0 += 32) {
    __syncthreads();                   // LDS WAR vs previous iter A-reads

    const float* xp = X + (size_t)(i0 + i_loc) * D1 + k0 + kl;
    v4f x0 = *(const v4f*)xp;
    v4f x1 = *(const v4f*)(xp + 4);
    if (i0 + 32 < NB) __builtin_prefetch(xp + (size_t)32 * D1, 0, 1); // global_prefetch_b8

    float tv = t_arr[i0 + i_loc];
    float xv[8] = {x0.x, x0.y, x0.z, x0.w, x1.x, x1.y, x1.z, x1.w};
    float sp = 0.f;
    #pragma unroll
    for (int q = 0; q < 8; ++q) {
      float x = xv[q];
      sp      += x * x;
      vacc[q] += x;
      wacc[q] += tv * x;
      __bf16 h  = (__bf16)x;
      __bf16 lo = (__bf16)(x - (float)h);
      Xh[(kl + q) * LDSROW + i_loc] = h;   // transposed store
      Xl[(kl + q) * LDSROW + i_loc] = lo;
    }
    // reduce s over the 8 lanes sharing row i, one atomic per (wg,row)
    sp += __shfl_xor(sp, 1, 32);
    sp += __shfl_xor(sp, 2, 32);
    sp += __shfl_xor(sp, 4, 32);
    if (j8 == 0) atomicAdd(&s_arr[i0 + i_loc], sp);

    __syncthreads();

    // A fragments (contiguous k order: lanes<16 hold K0..15; lanes>=16 K16..31, matching B)
    const __bf16* ah = &Xh[arow * LDSROW + half * 16];
    const __bf16* al = &Xl[arow * LDSROW + half * 16];
    v8bf ah0 = *(const v8bf*)ah;
    v8bf ah1 = *(const v8bf*)(ah + 8);
    v8bf al0 = *(const v8bf*)al;
    v8bf al1 = *(const v8bf*)(al + 8);
    v16bf Ah = __builtin_shufflevector(ah0, ah1, 0,1,2,3,4,5,6,7,8,9,10,11,12,13,14,15);
    v16bf Al = __builtin_shufflevector(al0, al1, 0,1,2,3,4,5,6,7,8,9,10,11,12,13,14,15);

    #pragma unroll
    for (int j = 0; j < 8; ++j) {
      const int l = (lgrp * 8 + j) * 16 + m;              // B col per lane
      const size_t boff = (size_t)l * NB + i0 + half * 16; // K range per lane half
      v16bf Bh = *(const v16bf*)(Nth + boff);              // 32B contiguous, L2-resident
      v16bf Bl = *(const v16bf*)(Ntl + boff);
      acc[j] = __builtin_amdgcn_wmma_f32_16x16x32_bf16(false, Ah, false, Bh, (short)0, acc[j], false, false);
      acc[j] = __builtin_amdgcn_wmma_f32_16x16x32_bf16(false, Ah, false, Bl, (short)0, acc[j], false, false);
      acc[j] = __builtin_amdgcn_wmma_f32_16x16x32_bf16(false, Al, false, Bh, (short)0, acc[j], false, false);
    }
  }

  // F += sum of squares of this wave's M-block
  float fs = 0.f;
  #pragma unroll
  for (int j = 0; j < 8; ++j) {
    #pragma unroll
    for (int r = 0; r < 8; ++r) fs += acc[j][r] * acc[j][r];
  }
  fs += __shfl_xor(fs, 16, 32);
  fs += __shfl_xor(fs, 8, 32);
  fs += __shfl_xor(fs, 4, 32);
  fs += __shfl_xor(fs, 2, 32);
  fs += __shfl_xor(fs, 1, 32);
  if (lane == 0) atomicAdd(&scal[0], fs);

  // v,w: combine across threads, then Svw += sum_k v*w for this slab
  #pragma unroll
  for (int q = 0; q < 8; ++q) {
    atomicAdd(&v_l[kl + q], vacc[q]);
    atomicAdd(&w_l[kl + q], wacc[q]);
  }
  __syncthreads();
  if (tid < KT) {
    float p = v_l[tid] * w_l[tid];
    p += __shfl_xor(p, 16, 32);
    p += __shfl_xor(p, 8, 32);
    p += __shfl_xor(p, 4, 32);
    p += __shfl_xor(p, 2, 32);
    p += __shfl_xor(p, 1, 32);
    if (lane == 0) atomicAdd(&scal[1], p);
  }
}

// ---- final scalar combine (fp64) ----
__global__ __launch_bounds__(256) void k_final(const float* __restrict__ s_arr,
                                               const float* __restrict__ t_arr,
                                               const float* __restrict__ rn,
                                               const float* __restrict__ scal,
                                               float* __restrict__ out) {
  __shared__ double red[4][256];
  int tid = threadIdx.x;
  double a_s = 0, a_t = 0, a_st = 0, a_srn = 0;
  for (int i = tid; i < NB; i += 256) {
    double si = s_arr[i], ti = t_arr[i], ri = rn[i];
    a_s += si; a_t += ti; a_st += si * ti; a_srn += si * ri;
  }
  red[0][tid] = a_s; red[1][tid] = a_t; red[2][tid] = a_st; red[3][tid] = a_srn;
  __syncthreads();
  for (int str = 128; str > 0; str >>= 1) {
    if (tid < str) {
      red[0][tid] += red[0][tid + str];
      red[1][tid] += red[1][tid + str];
      red[2][tid] += red[2][tid + str];
      red[3][tid] += red[3][tid + str];
    }
    __syncthreads();
  }
  if (tid == 0) {
    double Ss = red[0][0], St = red[1][0], Sst = red[2][0], Ssrn = red[3][0];
    double F = (double)scal[0], Svw = (double)scal[1];
    double total = 2.0 * NB * Sst + 2.0 * Ss * St - 4.0 * Ssrn - 4.0 * Svw + 4.0 * F;
    double mean = total / ((double)NB * (double)NB * (double)D1 * (double)D2);
    out[0] = (float)exp(-mean);
  }
}

extern "C" void kernel_launch(void* const* d_in, const int* in_sizes, int n_in,
                              void* d_out, int out_size, void* d_ws, size_t ws_size,
                              hipStream_t stream) {
  (void)in_sizes; (void)n_in; (void)out_size; (void)ws_size;
  const float* noises = (const float*)d_in[0];   // [4096,256]
  const float* images = (const float*)d_in[1];   // [4096,12288]

  char* base = (char*)d_ws;                      // needs ~4.25 MB
  __bf16* Nth  = (__bf16*)base;                                  // 2 MB
  __bf16* Ntl  = (__bf16*)(base + (size_t)2 * 1024 * 1024);      // 2 MB
  float*  t_arr = (float*)(base + (size_t)4 * 1024 * 1024);      // 16 KB
  float*  s_arr = t_arr + NB;                                    // 16 KB (accum)
  float*  rn    = t_arr + 2 * NB;                                // 16 KB
  float*  u     = t_arr + 3 * NB;                                // 1 KB (accum)
  float*  scal  = u + D2;                                        // [0]=F, [1]=Svw

  hipMemsetAsync(s_arr, 0, NB * sizeof(float), stream);
  hipMemsetAsync(u, 0, (D2 + 2) * sizeof(float), stream);        // u + scal

  k_noise_rowstats<<<NB / 256, 256, 0, stream>>>(noises, t_arr);
  k_noise_split   <<<(NB * D2) / 256, 256, 0, stream>>>(noises, Nth, Ntl);
  k_noise_colsum  <<<NB / 256, 256, 0, stream>>>(noises, u);
  k_noise_rn      <<<NB / 256, 256, 0, stream>>>(noises, u, rn);
  k_main          <<<D1 / KT, 256, 0, stream>>>(images, Nth, Ntl, t_arr, s_arr, scal);
  k_final         <<<1, 256, 0, stream>>>(s_arr, t_arr, rn, scal, (float*)d_out);
}
